// MultiSimilarityLoss_53489522704962
// MI455X (gfx1250) — matrix operand order repaired and checked by reading, not compiled, not run on hardware
//
#include <hip/hip_runtime.h>
#include <cstdint>
#include <cstddef>

typedef __attribute__((ext_vector_type(16))) _Float16 v16h;
typedef __attribute__((ext_vector_type(8)))  float    v8f;

constexpr int   NB = 8192;     // batch
constexpr int   ND = 1024;     // feature dim
constexpr float C_ALPHA  = 2.0f;
constexpr float C_BETA   = 50.0f;
constexpr float C_BASE   = 0.5f;
constexpr float C_MARGIN = 0.1f;

// ---------------------------------------------------------------- reductions
__device__ inline float blockReduce256(float v, float* sm) {
#pragma unroll
    for (int off = 16; off > 0; off >>= 1) v += __shfl_xor(v, off, 32);
    const int lane = threadIdx.x & 31;
    const int w    = threadIdx.x >> 5;
    if (lane == 0) sm[w] = v;
    __syncthreads();
    float r = 0.0f;
    if (threadIdx.x == 0) {
#pragma unroll
        for (int i = 0; i < 8; ++i) r += sm[i];
        sm[0] = r;
    }
    __syncthreads();
    r = sm[0];
    __syncthreads();   // safe reuse of sm by the next call
    return r;
}

// ------------------------------------------------- kernel 1: normalize + diag
__global__ __launch_bounds__(256)
void norm_rows(const float* __restrict__ f1, const float* __restrict__ f2,
               _Float16* __restrict__ f1hi, _Float16* __restrict__ f1lo,
               _Float16* __restrict__ f2hi, _Float16* __restrict__ f2lo,
               float* __restrict__ pos,
               float* __restrict__ rowsum, float* __restrict__ colsum) {
    const int row = blockIdx.x;
    const int tid = threadIdx.x;
    const size_t base = (size_t)row * ND + (size_t)tid * 4;

    const float4 a = *reinterpret_cast<const float4*>(f1 + base);
    const float4 b = *reinterpret_cast<const float4*>(f2 + base);

    float ss1 = a.x*a.x + a.y*a.y + a.z*a.z + a.w*a.w;
    float ss2 = b.x*b.x + b.y*b.y + b.z*b.z + b.w*b.w;
    float d12 = a.x*b.x + a.y*b.y + a.z*b.z + a.w*b.w;

    __shared__ float sm[8];
    ss1 = blockReduce256(ss1, sm);
    ss2 = blockReduce256(ss2, sm);
    d12 = blockReduce256(d12, sm);

    const float inv1 = rsqrtf(ss1);
    const float inv2 = rsqrtf(ss2);

    const float va[4] = {a.x*inv1, a.y*inv1, a.z*inv1, a.w*inv1};
    const float vb[4] = {b.x*inv2, b.y*inv2, b.z*inv2, b.w*inv2};
#pragma unroll
    for (int e = 0; e < 4; ++e) {
        _Float16 h1 = (_Float16)va[e];
        _Float16 h2 = (_Float16)vb[e];
        f1hi[base + e] = h1;
        f1lo[base + e] = (_Float16)(va[e] - (float)h1);
        f2hi[base + e] = h2;
        f2lo[base + e] = (_Float16)(vb[e] - (float)h2);
    }
    if (tid == 0) {
        pos[row]    = d12 * inv1 * inv2;
        rowsum[row] = 0.0f;
        colsum[row] = 0.0f;
    }
}

// --------------------------------------------- kernel 2: fused GEMM + mining
// Block tile 128x64, K-step 32, double-buffered LDS filled by CDNA5 async
// memory->LDS copies (GLOBAL_LOAD_ASYNC_TO_LDS_B128, ASYNCcnt-tracked).
// Last K-iteration peeled so the steady-state loop is branch-free.
__global__ __launch_bounds__(256)
void ms_gemm(const _Float16* __restrict__ f1hi, const _Float16* __restrict__ f1lo,
             const _Float16* __restrict__ f2hi, const _Float16* __restrict__ f2lo,
             const float* __restrict__ pos,
             float* __restrict__ rowsum, float* __restrict__ colsum) {
    constexpr int BM = 128, BN = 64, BK = 32, LP = 40;  // LP: padded LDS stride

    __shared__ alignas(16) _Float16 sAhi[2][BM][LP];
    __shared__ alignas(16) _Float16 sAlo[2][BM][LP];
    __shared__ alignas(16) _Float16 sBhi[2][BN][LP];
    __shared__ alignas(16) _Float16 sBlo[2][BN][LP];
    __shared__ float rs[BM], cs[BN], pr[BM], pc[BN];

    const int tid     = threadIdx.x;
    const int rowBase = blockIdx.y * BM;
    const int colBase = blockIdx.x * BN;

    if (tid < BM) { rs[tid] = 0.0f; pr[tid] = pos[rowBase + tid]; }
    if (tid < BN) { cs[tid] = 0.0f; pc[tid] = pos[colBase + tid]; }

    const int wave = tid >> 5, lane = tid & 31;
    const int wm = wave & 3, wn = wave >> 2;       // 4x2 wave grid
    const int g  = lane >> 4, mn = lane & 15;      // half-wave group / lane id

    // cooperative staging indices
    const int arow = tid >> 1, acol = (tid & 1) * 16;   // A: 128x32 halves
    const int brow = tid >> 2, bcol = (tid & 3) * 8;    // B: 64x32 halves

    // CDNA5 async global->LDS B128 copy (no VGPR round-trip, ASYNCcnt-tracked)
    auto async_cp16 = [](const _Float16* sdst, const _Float16* gsrc) {
        const unsigned lds = (unsigned)(uintptr_t)sdst;   // addr[31:0] = LDS offset
        asm volatile("global_load_async_to_lds_b128 %0, %1, off"
                     :: "v"(lds), "v"((unsigned long long)(uintptr_t)gsrc)
                     : "memory");
    };
    auto stage = [&](int buf, int k0) {
        const size_t ga = (size_t)(rowBase + arow) * ND + k0 + acol;
        async_cp16(&sAhi[buf][arow][acol],     f1hi + ga);
        async_cp16(&sAhi[buf][arow][acol + 8], f1hi + ga + 8);
        async_cp16(&sAlo[buf][arow][acol],     f1lo + ga);
        async_cp16(&sAlo[buf][arow][acol + 8], f1lo + ga + 8);
        const size_t gb = (size_t)(colBase + brow) * ND + k0 + bcol;
        async_cp16(&sBhi[buf][brow][bcol], f2hi + gb);
        async_cp16(&sBlo[buf][brow][bcol], f2lo + gb);
    };

    v8f c[2][2] = {};

    // fragment loads (ISA 16-bit A layout: halves 0-7=K[g*8..], 8-15=K[16+g*8..];
    //                 16-bit B layout: lane=N, halves 0-7=K[g*8..], 8-15=K[16+g*8..])
    auto compute = [&](int buf) {
        v16h ahi[2], alo[2], bhi[2], blo[2];
#pragma unroll
        for (int tm = 0; tm < 2; ++tm) {
            const int rA = wm * 32 + tm * 16 + mn;
            uint4* p = reinterpret_cast<uint4*>(&ahi[tm]);
            p[0] = *reinterpret_cast<const uint4*>(&sAhi[buf][rA][g * 8]);
            p[1] = *reinterpret_cast<const uint4*>(&sAhi[buf][rA][16 + g * 8]);
            uint4* q = reinterpret_cast<uint4*>(&alo[tm]);
            q[0] = *reinterpret_cast<const uint4*>(&sAlo[buf][rA][g * 8]);
            q[1] = *reinterpret_cast<const uint4*>(&sAlo[buf][rA][16 + g * 8]);
        }
#pragma unroll
        for (int tn = 0; tn < 2; ++tn) {
            const int rB = wn * 32 + tn * 16 + mn;
            uint4* p = reinterpret_cast<uint4*>(&bhi[tn]);
            p[0] = *reinterpret_cast<const uint4*>(&sBhi[buf][rB][g * 8]);
            p[1] = *reinterpret_cast<const uint4*>(&sBhi[buf][rB][16 + g * 8]);
            uint4* q = reinterpret_cast<uint4*>(&blo[tn]);
            q[0] = *reinterpret_cast<const uint4*>(&sBlo[buf][rB][g * 8]);
            q[1] = *reinterpret_cast<const uint4*>(&sBlo[buf][rB][16 + g * 8]);
        }
#pragma unroll
        for (int tm = 0; tm < 2; ++tm)
#pragma unroll
            for (int tn = 0; tn < 2; ++tn) {
                c[tm][tn] = __builtin_amdgcn_wmma_f32_16x16x32_f16(
                    false, ahi[tm], false, bhi[tn], (short)0, c[tm][tn], false, false);
                c[tm][tn] = __builtin_amdgcn_wmma_f32_16x16x32_f16(
                    false, ahi[tm], false, blo[tn], (short)0, c[tm][tn], false, false);
                c[tm][tn] = __builtin_amdgcn_wmma_f32_16x16x32_f16(
                    false, alo[tm], false, bhi[tn], (short)0, c[tm][tn], false, false);
            }
    };

    stage(0, 0);                               // prime buffer 0 (6 async ops)
    int ib = 0;
    // steady state: always prefetch -> branch-free pipeline
    for (int k0 = 0; k0 < ND - BK; k0 += BK, ib ^= 1) {
        __syncthreads();                       // all waves done READING buf ib^1
        stage(ib ^ 1, k0 + BK);                // prefetch next (6 more async)
        asm volatile("s_wait_asynccnt 0x6" ::: "memory");   // buf ib landed
        __syncthreads();                       // buf ib visible to all waves
        compute(ib);
    }
    // peeled tail: no prefetch, drain everything
    __syncthreads();
    asm volatile("s_wait_asynccnt 0x0" ::: "memory");
    __syncthreads();
    compute(ib);

    // epilogue: C/D layout -> M = r + 8*g, N = mn (per 16x16 f32 table)
    float cAcc[2] = {0.0f, 0.0f};
    int jloc[2];
#pragma unroll
    for (int tn = 0; tn < 2; ++tn) jloc[tn] = wn * 32 + tn * 16 + mn;

#pragma unroll
    for (int tm = 0; tm < 2; ++tm) {
#pragma unroll
        for (int r = 0; r < 8; ++r) {
            const int li = wm * 32 + tm * 16 + 8 * g + r;   // local row
            const int i  = rowBase + li;
            float rAcc = 0.0f;
#pragma unroll
            for (int tn = 0; tn < 2; ++tn) {
                const float s = c[tm][tn][r];
                const int   j = colBase + jloc[tn];
                if (i != j) {
                    const float e = __expf(C_BETA * (s - C_BASE));
                    if (s + C_MARGIN > pr[li])       rAcc     += e;  // i2t row view
                    if (s + C_MARGIN > pc[jloc[tn]]) cAcc[tn] += e;  // t2i col view
                }
            }
            atomicAdd(&rs[li], rAcc);
        }
    }
#pragma unroll
    for (int tn = 0; tn < 2; ++tn) atomicAdd(&cs[jloc[tn]], cAcc[tn]);

    __syncthreads();
    if (tid < BM) atomicAdd(&rowsum[rowBase + tid], rs[tid]);
    if (tid < BN) atomicAdd(&colsum[colBase + tid], cs[tid]);
}

// -------------------------------------------------------- kernel 3: finalize
__global__ __launch_bounds__(256)
void finalize(const float* __restrict__ pos, const float* __restrict__ rowsum,
              const float* __restrict__ colsum, float* __restrict__ out) {
    const int tid = threadIdx.x;
    float aP = 0.0f, aR = 0.0f, aC = 0.0f;
    for (int i = tid; i < NB; i += 256) {
        const float p = pos[i];
        aP += log1pf(__expf(-C_ALPHA * (p - C_BASE))) / C_ALPHA;  // softplus
        aR += log1pf(rowsum[i]);
        aC += log1pf(colsum[i]);
    }
    __shared__ float sm[8];
    aP = blockReduce256(aP, sm);
    aR = blockReduce256(aR, sm);
    aC = blockReduce256(aC, sm);
    if (tid == 0)
        out[0] = aP / (float)NB + (aR + aC) / (2.0f * C_BETA * (float)NB);
}

// ------------------------------------------------------------------- launch
extern "C" void kernel_launch(void* const* d_in, const int* in_sizes, int n_in,
                              void* d_out, int out_size, void* d_ws, size_t ws_size,
                              hipStream_t stream) {
    const float* f1 = (const float*)d_in[0];
    const float* f2 = (const float*)d_in[1];
    float* out = (float*)d_out;

    const size_t NE = (size_t)NB * ND;
    _Float16* f1hi = (_Float16*)d_ws;
    _Float16* f1lo = f1hi + NE;
    _Float16* f2hi = f1lo + NE;
    _Float16* f2lo = f2hi + NE;
    float* pos    = (float*)(f2lo + NE);
    float* rowsum = pos + NB;
    float* colsum = rowsum + NB;

    norm_rows<<<NB, 256, 0, stream>>>(f1, f2, f1hi, f1lo, f2hi, f2lo,
                                      pos, rowsum, colsum);

    dim3 grid(NB / 64, NB / 128);   // x: col blocks (BN=64), y: row blocks (BM=128)
    ms_gemm<<<grid, 256, 0, stream>>>(f1hi, f1lo, f2hi, f2lo,
                                      pos, rowsum, colsum);

    finalize<<<1, 256, 0, stream>>>(pos, rowsum, colsum, out);
}
